// SparseBiLevel_80685255623316
// MI455X (gfx1250) — matrix operand order repaired and checked by reading, not compiled, not run on hardware
//
#include <hip/hip_runtime.h>
#include <math.h>

typedef __attribute__((ext_vector_type(16))) __bf16 v16bf;
typedef __attribute__((ext_vector_type(8)))  float  v8f;

#define N_ROWS 131072
#define D_DIM  1024
#define NWAYS  256
#define KSLICE 2048   // GEMM2 K-slice; 131072/2048 = 64 slices

typedef unsigned short u16;
typedef unsigned int   u32;

// Scheduling-group pin: keep prefetch loads issued ahead of the WMMA that
// overlaps them (mask 0x20 = VMEM read, 0x8 = MFMA/WMMA).
#if defined(__has_builtin)
#  if __has_builtin(__builtin_amdgcn_sched_group_barrier)
#    define SGB(mask, cnt, sync) __builtin_amdgcn_sched_group_barrier(mask, cnt, sync)
#  else
#    define SGB(mask, cnt, sync)
#  endif
#else
#  define SGB(mask, cnt, sync)
#endif

// ---------- helpers ----------
__device__ __forceinline__ u16 f2bf(float f) {
    u32 u = __float_as_uint(f);
    u += 0x7FFFu + ((u >> 16) & 1u);   // round-to-nearest-even
    return (u16)(u >> 16);
}

// A-fragment (16x32 bf16): lane half splits K 0-7/16-23 vs 8-15/24-31
__device__ __forceinline__ v16bf load_a_frag(const u16* __restrict__ row, int k0, int khalf) {
    union { uint4 q[2]; v16bf v; } u;
    u.q[0] = *(const uint4*)(row + k0 + 8 * khalf);
    u.q[1] = *(const uint4*)(row + k0 + 16 + 8 * khalf);
    return u.v;
}

// B-fragment (32x16 bf16): lane half splits K 0-7/16-23 vs 8-15/24-31
__device__ __forceinline__ v16bf load_b_frag(const u16* __restrict__ col, int k0, int khalf) {
    union { uint4 q[2]; v16bf v; } u;
    u.q[0] = *(const uint4*)(col + k0 + 8 * khalf);
    u.q[1] = *(const uint4*)(col + k0 + 16 + 8 * khalf);
    return u.v;
}

// ---------- K0: zero W, m, v, g, WT ----------
__global__ void __launch_bounds__(256) init_kernel(
        float* __restrict__ W, float* __restrict__ m,
        float* __restrict__ v, float* __restrict__ g,
        u16* __restrict__ WT) {
    int id = blockIdx.x * 256 + threadIdx.x;   // 262144 total
    W[id] = 0.0f; m[id] = 0.0f; v[id] = 0.0f; g[id] = 0.0f; WT[id] = 0;
}

// ---------- K1: fp32 X -> bf16 Xb (row-major) + XT (transposed), LDS 32x32 tiles ----------
__global__ void __launch_bounds__(256) convert_kernel(
        const float* __restrict__ X,
        u16* __restrict__ Xb, u16* __restrict__ XT) {
    __shared__ u16 lds[32 * 33];
    int tx = threadIdx.x & 31;          // d within tile
    int ty = threadIdx.x >> 5;          // 0..7
    int bx = blockIdx.x & 31;           // d-block (1024/32)
    int by = blockIdx.x >> 5;           // n-block (131072/32)
    int d  = bx * 32 + tx;
    #pragma unroll
    for (int rr = 0; rr < 4; ++rr) {
        int r = ty + rr * 8;
        int n = by * 32 + r;
        u16 h = f2bf(X[(size_t)n * D_DIM + d]);
        Xb[(size_t)n * D_DIM + d] = h;          // coalesced
        lds[r * 33 + tx] = h;
    }
    __syncthreads();
    #pragma unroll
    for (int rr = 0; rr < 4; ++rr) {
        int r = ty + rr * 8;                    // d within tile for output
        XT[(size_t)(bx * 32 + r) * N_ROWS + by * 32 + tx] = lds[tx * 33 + r]; // coalesced
    }
}

// ---------- K2: fused logits = Xb @ W, row softmax, PT[w][n] = (p - onehot)/N (bf16) ----------
// One wave owns a 16-row strip x all 256 ways (16 accumulator tiles).
// B-frags issued one stage ahead; sched_group_barrier pins (2 loads, 1 wmma)
// per stage so the scheduler cannot sink the prefetch below the WMMA.
__global__ void __launch_bounds__(128, 1) fwd_softmax_kernel(
        const u16* __restrict__ Xb, const u16* __restrict__ WT,
        const int* __restrict__ targets, u16* __restrict__ PT) {
    int lane  = threadIdx.x & 31;
    int wave  = threadIdx.x >> 5;
    int strip = blockIdx.x * 4 + wave;      // 8192 strips of 16 rows
    int khalf = lane >> 4;
    int ln    = lane & 15;

    const u16* arow  = Xb + (size_t)(strip * 16 + ln) * D_DIM;
    const u16* bbase = WT + (size_t)ln * D_DIM;

    v8f acc[16];
    v8f zero = {0.f, 0.f, 0.f, 0.f, 0.f, 0.f, 0.f, 0.f};
    #pragma unroll
    for (int t = 0; t < 16; ++t) acc[t] = zero;

    v16bf a = load_a_frag(arow, 0, khalf);
    v16bf b = load_b_frag(bbase, 0, khalf);
    for (int k0 = 0; k0 < D_DIM; k0 += 32) {
        // Prefetch next chunk's A (one-chunk overrun lands in XT: safe)
        v16bf a_next = load_a_frag(arow, k0 + 32, khalf);
        SGB(0x20, 2, 0);                       // 2 VMEM reads (A prefetch)
        #pragma unroll
        for (int t = 0; t < 16; ++t) {
            v16bf b_next;
            if (t < 15)
                b_next = load_b_frag(bbase + (size_t)(t + 1) * 16 * D_DIM, k0, khalf);
            else
                b_next = load_b_frag(bbase, k0 + 32, khalf);  // next k, t=0 (over-read stays inside WT: safe)
            acc[t] = __builtin_amdgcn_wmma_f32_16x16x32_bf16(
                false, a, false, b, (short)0, acc[t], false, false);
            SGB(0x20, 2, 0);                   // 2 VMEM reads (B prefetch)
            SGB(0x8, 1, 0);                    // 1 WMMA
            b = b_next;
        }
        a = a_next;
    }

    // --- row softmax: row r lives in VGPR r across lanes {khalf*16 .. +15} x 16 tiles ---
    float mx[8], sm[8];
    #pragma unroll
    for (int r = 0; r < 8; ++r) mx[r] = -3.402823e38f;
    #pragma unroll
    for (int t = 0; t < 16; ++t)
        #pragma unroll
        for (int r = 0; r < 8; ++r) mx[r] = fmaxf(mx[r], acc[t][r]);
    #pragma unroll
    for (int msk = 1; msk <= 8; msk <<= 1)
        #pragma unroll
        for (int r = 0; r < 8; ++r) mx[r] = fmaxf(mx[r], __shfl_xor(mx[r], msk, 32));
    #pragma unroll
    for (int r = 0; r < 8; ++r) sm[r] = 0.f;
    #pragma unroll
    for (int t = 0; t < 16; ++t)
        #pragma unroll
        for (int r = 0; r < 8; ++r) {
            float e = __expf(acc[t][r] - mx[r]);
            acc[t][r] = e;
            sm[r] += e;
        }
    #pragma unroll
    for (int msk = 1; msk <= 8; msk <<= 1)
        #pragma unroll
        for (int r = 0; r < 8; ++r) sm[r] += __shfl_xor(sm[r], msk, 32);

    int roff  = khalf * 8;
    int nbase = strip * 16 + roff;
    int tg[8];
    #pragma unroll
    for (int r = 0; r < 8; ++r) tg[r] = targets[nbase + r];

    const float invN = 1.0f / (float)N_ROWS;
    #pragma unroll
    for (int t = 0; t < 16; ++t) {
        int col = t * 16 + ln;
        u16 h[8];
        #pragma unroll
        for (int r = 0; r < 8; ++r) {
            float p = acc[t][r] / sm[r];
            p = (p - ((tg[r] == col) ? 1.0f : 0.0f)) * invN;
            h[r] = f2bf(p);
        }
        uint4 q;
        q.x = (u32)h[0] | ((u32)h[1] << 16);
        q.y = (u32)h[2] | ((u32)h[3] << 16);
        q.z = (u32)h[4] | ((u32)h[5] << 16);
        q.w = (u32)h[6] | ((u32)h[7] << 16);
        *(uint4*)(PT + (size_t)col * N_ROWS + nbase) = q;   // 16B packed store
    }
}

// ---------- K3: g[d][w] += XT-strip @ PT-slice (64x64 per wave, K-sliced, atomic reduce) ----------
// Double-buffered: B across j-loop, A for next k-chunk prefetched inside j-loop;
// sched_group_barrier pins (4 loads, 4 wmma) per j-stage.
__global__ void __launch_bounds__(32, 1) grad_kernel(
        const u16* __restrict__ XT, const u16* __restrict__ PT,
        float* __restrict__ g) {
    int lane  = threadIdx.x;            // blockDim = 32
    int khalf = lane >> 4;
    int ln    = lane & 15;
    int bid   = blockIdx.x;             // kslice slowest -> L2 captures the slice
    int kslc  = bid >> 6;               // 64 slices
    int mb    = (bid >> 2) & 15;        // 16 m-blocks of 64 rows (d)
    int nb    = bid & 3;                // 4 n-blocks of 64 cols (w)
    int kbeg  = kslc * KSLICE;
    int kend  = kbeg + KSLICE;

    const u16* abase = XT + (size_t)(mb * 64 + ln) * N_ROWS;
    const u16* bbase = PT + (size_t)(nb * 64 + ln) * N_ROWS;

    v8f acc[16];
    v8f zero = {0.f, 0.f, 0.f, 0.f, 0.f, 0.f, 0.f, 0.f};
    #pragma unroll
    for (int t = 0; t < 16; ++t) acc[t] = zero;

    v16bf a[4];
    #pragma unroll
    for (int i = 0; i < 4; ++i)
        a[i] = load_a_frag(abase + (size_t)i * 16 * N_ROWS, kbeg, khalf);
    v16bf b = load_b_frag(bbase, kbeg, khalf);

    for (int k = kbeg; k < kend; k += 32) {
        v16bf a_next[4];
        #pragma unroll
        for (int j = 0; j < 4; ++j) {
            // Prefetch one A-frag for the next chunk (overruns land in PT: safe)
            a_next[j] = load_a_frag(abase + (size_t)j * 16 * N_ROWS, k + 32, khalf);
            v16bf b_next;
            if (j < 3)
                b_next = load_b_frag(bbase + (size_t)(j + 1) * 16 * N_ROWS, k, khalf);
            else
                b_next = load_b_frag(bbase, k + 32, khalf);   // next k, j=0
            #pragma unroll
            for (int i = 0; i < 4; ++i)
                acc[i * 4 + j] = __builtin_amdgcn_wmma_f32_16x16x32_bf16(
                    false, a[i], false, b, (short)0, acc[i * 4 + j], false, false);
            SGB(0x20, 4, 0);               // 4 VMEM reads (A + B prefetch)
            SGB(0x8, 4, 0);                // 4 WMMAs
            b = b_next;
        }
        #pragma unroll
        for (int i = 0; i < 4; ++i) a[i] = a_next[i];
    }

    int roff = khalf * 8;
    #pragma unroll
    for (int i = 0; i < 4; ++i)
        #pragma unroll
        for (int j = 0; j < 4; ++j)
            #pragma unroll
            for (int r = 0; r < 8; ++r)
                atomicAdd(&g[(size_t)(mb * 64 + i * 16 + roff + r) * NWAYS
                             + nb * 64 + j * 16 + ln],
                          acc[i * 4 + j][r]);
}

// ---------- K4: Adam update + refresh bf16 WT, zero g for next step ----------
__global__ void __launch_bounds__(256) adam_kernel(
        float* __restrict__ W, float* __restrict__ m,
        float* __restrict__ v, float* __restrict__ g,
        u16* __restrict__ WT, float c1, float c2) {
    int id = blockIdx.x * 256 + threadIdx.x;    // 262144 = D*WAYS, row-major (d,w)
    float gv = g[id];
    g[id] = 0.0f;
    float mm = 0.9f * m[id] + 0.1f * gv;
    float vv = 0.999f * v[id] + 0.001f * gv * gv;
    m[id] = mm; v[id] = vv;
    float w = W[id] - 0.01f * (mm * c1) / (sqrtf(vv * c2) + 1e-8f);
    W[id] = w;
    int d = id >> 8, way = id & 255;
    WT[(size_t)way * D_DIM + d] = f2bf(w);
}

// ---------- host ----------
extern "C" void kernel_launch(void* const* d_in, const int* in_sizes, int n_in,
                              void* d_out, int out_size, void* d_ws, size_t ws_size,
                              hipStream_t stream) {
    const float* X       = (const float*)d_in[0];
    const int*   targets = (const int*)d_in[1];
    (void)in_sizes; (void)n_in; (void)out_size; (void)ws_size;

    // Order matters: one-chunk-ahead prefetches may over-read up to 64B past a
    // matrix's last row; each buffer is followed by another ws buffer so the
    // overrun stays inside the workspace. g is last and never prefetched.
    char* ws = (char*)d_ws;
    const size_t SZ_XB = (size_t)N_ROWS * D_DIM * 2;         // 256 MB
    const size_t SZ_PT = (size_t)NWAYS * N_ROWS * 2;         // 64 MB
    const size_t SZ_MV = (size_t)D_DIM * NWAYS * 4;          // 1 MB
    const size_t SZ_WT = (size_t)D_DIM * NWAYS * 2;          // 512 KB
    u16*   Xb = (u16*)(ws);
    u16*   XT = (u16*)(ws + SZ_XB);
    u16*   PT = (u16*)(ws + 2 * SZ_XB);
    float* m  = (float*)(ws + 2 * SZ_XB + SZ_PT);
    float* v  = (float*)(ws + 2 * SZ_XB + SZ_PT + SZ_MV);
    u16*   WT = (u16*)(ws + 2 * SZ_XB + SZ_PT + 2 * SZ_MV);
    float* g  = (float*)(ws + 2 * SZ_XB + SZ_PT + 2 * SZ_MV + SZ_WT);
    float* W  = (float*)d_out;                               // [D][WAYS] fp32

    init_kernel<<<1024, 256, 0, stream>>>(W, m, v, g, WT);
    convert_kernel<<<32 * (N_ROWS / 32), 256, 0, stream>>>(X, Xb, XT);

    for (int t = 1; t <= 5; ++t) {
        fwd_softmax_kernel<<<(N_ROWS / 16) / 4, 128, 0, stream>>>(Xb, WT, targets, PT);
        grad_kernel<<<(N_ROWS / KSLICE) * 16 * 4, 32, 0, stream>>>(XT, PT, g);
        float c1 = 1.0f / (1.0f - powf(0.9f, (float)t));
        float c2 = 1.0f / (1.0f - powf(0.999f, (float)t));
        adam_kernel<<<1024, 256, 0, stream>>>(W, m, v, g, WT, c1, c2);
    }
}
